// VectorQuantizerEMA_72292889526318
// MI455X (gfx1250) — matrix operand order repaired and checked by reading, not compiled, not run on hardware
//
#include <hip/hip_runtime.h>
#include <hip/hip_bf16.h>
#include <stdint.h>
#include <stddef.h>

#define DECAY 0.99f
#define EPS 1e-5f
#define N_TOK 32768
#define DIM 256
#define K_EMB 4096
#define LDSP 272  // padded row stride in ushorts (544 bytes, 16B aligned rows)

#if defined(__has_builtin)
#  if __has_builtin(__builtin_amdgcn_global_load_async_to_lds_b128)
#    define ASYNC_LDS 1
#  endif
#endif
#ifndef ASYNC_LDS
#  define ASYNC_LDS 0
#endif

#define GAS __attribute__((address_space(1)))
#define LAS __attribute__((address_space(3)))

typedef __attribute__((ext_vector_type(16))) __bf16 v16bf;
typedef __attribute__((ext_vector_type(8)))  float  v8f;
typedef __attribute__((ext_vector_type(4)))  int    v4i;

union BF16x16 { v16bf v; unsigned short u[16]; uint4 q[2]; };

static __device__ inline unsigned short f2bf(float f) {
  union { float f; unsigned u; } v; v.f = f;
  unsigned r = v.u + 0x7FFFu + ((v.u >> 16) & 1u);   // round-to-nearest-even
  return (unsigned short)(r >> 16);
}
static __device__ inline float bf2f(unsigned short h) {
  union { unsigned u; float f; } v; v.u = ((unsigned)h) << 16;
  return v.f;
}
// monotonic float -> uint mapping (total order, min-compatible)
static __device__ inline unsigned ford(float f) {
  union { float f; unsigned u; } v; v.f = f;
  return (v.u & 0x80000000u) ? ~v.u : (v.u | 0x80000000u);
}
static __device__ inline unsigned long long shfl_xor_u64(unsigned long long v, int m) {
  unsigned lo = (unsigned)v, hi = (unsigned)(v >> 32);
  lo = __shfl_xor(lo, m, 32);
  hi = __shfl_xor(hi, m, 32);
  return ((unsigned long long)hi << 32) | lo;
}
static __device__ inline v8f wmma_bf16(v16bf a, v16bf b, v8f c) {
  return __builtin_amdgcn_wmma_f32_16x16x32_bf16(false, a, false, b, (short)0, c,
                                                 false, false);
}

// ---------------------------------------------------------------- zero fill
__global__ void fill0_kernel(float4* p, size_t n4) {
  size_t i = (size_t)blockIdx.x * blockDim.x + threadIdx.x;
  size_t stride = (size_t)gridDim.x * blockDim.x;
  float4 z = make_float4(0.f, 0.f, 0.f, 0.f);
  for (; i < n4; i += stride) p[i] = z;
}

// ------------------------------------------- codebook -> bf16 hi/lo + norms
// one wave per codebook row; lane handles 8 elements
__global__ __launch_bounds__(256) void prep_codebook_kernel(
    const float* __restrict__ cb, unsigned short* __restrict__ cb_hi,
    unsigned short* __restrict__ cb_lo, float* __restrict__ enorm) {
  int lane = threadIdx.x & 31;
  int k = blockIdx.x * 8 + (threadIdx.x >> 5);
  const float* row = cb + (size_t)k * DIM + lane * 8;
  float x[8];
  *(float4*)(x + 0) = *(const float4*)(row + 0);
  *(float4*)(x + 4) = *(const float4*)(row + 4);
  float s = 0.f;
  unsigned short h[8], l[8];
#pragma unroll
  for (int j = 0; j < 8; ++j) {
    s += x[j] * x[j];
    h[j] = f2bf(x[j]);
    l[j] = f2bf(x[j] - bf2f(h[j]));
  }
  uint4 ph, pl;
  ph.x = h[0] | (h[1] << 16); ph.y = h[2] | (h[3] << 16);
  ph.z = h[4] | (h[5] << 16); ph.w = h[6] | (h[7] << 16);
  pl.x = l[0] | (l[1] << 16); pl.y = l[2] | (l[3] << 16);
  pl.z = l[4] | (l[5] << 16); pl.w = l[6] | (l[7] << 16);
  *(uint4*)(cb_hi + (size_t)k * DIM + lane * 8) = ph;
  *(uint4*)(cb_lo + (size_t)k * DIM + lane * 8) = pl;
#pragma unroll
  for (int m = 16; m >= 1; m >>= 1) s += __shfl_xor(s, m, 32);
  if (lane == 0) enorm[k] = s;
}

// ----------------------------------------------------------- main VQ kernel
// 8 waves / WG; each wave owns TWO 16-token tiles (A-reuse halves LDS B-read
// per WMMA); WG streams 16-code tiles of bf16 hi/lo codebook through LDS.
__global__ __launch_bounds__(256) void vq_main_kernel(
    const float* __restrict__ X, const float* __restrict__ CB,
    const unsigned short* __restrict__ cb_hi,
    const unsigned short* __restrict__ cb_lo,
    const float* __restrict__ enorm,
    float* __restrict__ q_out, float* __restrict__ enc_out,
    float* __restrict__ dw, float* __restrict__ counts) {
  __shared__ unsigned short lds_hi[16 * LDSP];
  __shared__ unsigned short lds_lo[16 * LDSP];

  const int tid  = threadIdx.x;
  const int lane = tid & 31;
  const int wid  = tid >> 5;
  const int hgrp = lane >> 4;     // K-split half of the wave
  const int col  = lane & 15;     // N column / A row index
  const int m0   = (blockIdx.x * 8 + wid) * 32;   // 32 tokens per wave

  // --- build A fragments (bf16 hi/lo), resident in registers -------------
  BF16x16 ahi[2][8], alo[2][8];
#pragma unroll
  for (int tt = 0; tt < 2; ++tt) {
    const float* xrow = X + (size_t)(m0 + tt * 16 + col) * DIM;
#pragma unroll
    for (int s = 0; s < 8; ++s) {
      const int kc1 = s * 32 + hgrp * 8;        // K chunk 1 (8 values)
      const int kc2 = s * 32 + 16 + hgrp * 8;   // K chunk 2 (8 values)
      float xv[16];
      *(float4*)(xv + 0)  = *(const float4*)(xrow + kc1);
      *(float4*)(xv + 4)  = *(const float4*)(xrow + kc1 + 4);
      *(float4*)(xv + 8)  = *(const float4*)(xrow + kc2);
      *(float4*)(xv + 12) = *(const float4*)(xrow + kc2 + 4);
#pragma unroll
      for (int j = 0; j < 16; ++j) {
        unsigned short h = f2bf(xv[j]);
        ahi[tt][s].u[j] = h;
        alo[tt][s].u[j] = f2bf(xv[j] - bf2f(h));
      }
    }
  }

  unsigned long long best[2][8];
#pragma unroll
  for (int tt = 0; tt < 2; ++tt)
#pragma unroll
    for (int r = 0; r < 8; ++r) best[tt][r] = ~0ull;

  // --- stream codebook tiles --------------------------------------------
  for (int t = 0; t < K_EMB / 16; ++t) {
    __syncthreads();
    {  // cooperative fill: 16 rows x 256 bf16 (hi + lo)
      const int rr = tid >> 4;   // row 0..15
      const int cc = tid & 15;   // 16-ushort chunk 0..15
      const unsigned short* gh = cb_hi + (size_t)(t * 16 + rr) * DIM + cc * 16;
      const unsigned short* gl = cb_lo + (size_t)(t * 16 + rr) * DIM + cc * 16;
      unsigned short* lh = lds_hi + rr * LDSP + cc * 16;
      unsigned short* ll = lds_lo + rr * LDSP + cc * 16;
#if ASYNC_LDS
      __builtin_amdgcn_global_load_async_to_lds_b128((GAS v4i*)gh,       (LAS v4i*)lh,       0, 0);
      __builtin_amdgcn_global_load_async_to_lds_b128((GAS v4i*)(gh + 8), (LAS v4i*)(lh + 8), 0, 0);
      __builtin_amdgcn_global_load_async_to_lds_b128((GAS v4i*)gl,       (LAS v4i*)ll,       0, 0);
      __builtin_amdgcn_global_load_async_to_lds_b128((GAS v4i*)(gl + 8), (LAS v4i*)(ll + 8), 0, 0);
#  if defined(__has_builtin) && __has_builtin(__builtin_amdgcn_s_wait_asynccnt)
      __builtin_amdgcn_s_wait_asynccnt(0);
#  else
      asm volatile("s_wait_asynccnt 0x0" ::: "memory");
#  endif
#else
      ((uint4*)lh)[0] = ((const uint4*)gh)[0];
      ((uint4*)lh)[1] = ((const uint4*)gh)[1];
      ((uint4*)ll)[0] = ((const uint4*)gl)[0];
      ((uint4*)ll)[1] = ((const uint4*)gl)[1];
#endif
    }
    __syncthreads();

    v8f acc0 = {0.f, 0.f, 0.f, 0.f, 0.f, 0.f, 0.f, 0.f};
    v8f acc1 = {0.f, 0.f, 0.f, 0.f, 0.f, 0.f, 0.f, 0.f};
#pragma unroll
    for (int s = 0; s < 8; ++s) {
      const int koff = s * 32 + hgrp * 8;  // B: K chunks koff..+7 and koff+16..+23
      BF16x16 bh, bl;
      const uint4* ph = (const uint4*)(lds_hi + col * LDSP + koff);
      const uint4* pl = (const uint4*)(lds_lo + col * LDSP + koff);
      bh.q[0] = ph[0]; bh.q[1] = ph[2];
      bl.q[0] = pl[0]; bl.q[1] = pl[2];
      // 6 WMMAs per B-fragment pair (B reused across two token tiles)
      acc0 = wmma_bf16(ahi[0][s].v, bh.v, acc0);
      acc1 = wmma_bf16(ahi[1][s].v, bh.v, acc1);
      acc0 = wmma_bf16(ahi[0][s].v, bl.v, acc0);
      acc1 = wmma_bf16(ahi[1][s].v, bl.v, acc1);
      acc0 = wmma_bf16(alo[0][s].v, bh.v, acc0);
      acc1 = wmma_bf16(alo[1][s].v, bh.v, acc1);
    }

    const int n = t * 16 + col;
    const float en = enorm[n];
#pragma unroll
    for (int r = 0; r < 8; ++r) {
      // argmin of ||e||^2 - 2 x.e  (||x||^2 is row-constant)
      float s0 = en - 2.0f * acc0[r];
      unsigned long long k0 =
          ((unsigned long long)ford(s0) << 32) | (unsigned)n;
      best[0][r] = k0 < best[0][r] ? k0 : best[0][r];
      float s1 = en - 2.0f * acc1[r];
      unsigned long long k1 =
          ((unsigned long long)ford(s1) << 32) | (unsigned)n;
      best[1][r] = k1 < best[1][r] ? k1 : best[1][r];
    }
  }

  // --- reduce argmin across the 16-lane half-groups ----------------------
#pragma unroll
  for (int tt = 0; tt < 2; ++tt) {
#pragma unroll
    for (int r = 0; r < 8; ++r) {
#pragma unroll
      for (int m = 1; m < 16; m <<= 1) {
        unsigned long long o = shfl_xor_u64(best[tt][r], m);
        best[tt][r] = o < best[tt][r] ? o : best[tt][r];
      }
    }
  }

  // --- outputs: one-hot, quantized gather, dw/counts scatter -------------
#pragma unroll
  for (int tt = 0; tt < 2; ++tt) {
#pragma unroll
    for (int r = 0; r < 8; ++r) {
      const int token = m0 + tt * 16 + r + 8 * hgrp;  // C/D layout: M=r+8*half
      const unsigned idx = (unsigned)(best[tt][r] & 0xFFFFFFFFu);
      const float* crow = CB + (size_t)idx * DIM + col * 16;
      const float* xr   = X  + (size_t)token * DIM + col * 16;
      float* qr = q_out + (size_t)token * DIM + col * 16;
      float* dr = dw    + (size_t)idx * DIM + col * 16;
#pragma unroll
      for (int j = 0; j < 16; j += 4) {
        float4 cv = *(const float4*)(crow + j);
        *(float4*)(qr + j) = cv;
        float4 xv = *(const float4*)(xr + j);
        atomicAdd(dr + j + 0, xv.x);
        atomicAdd(dr + j + 1, xv.y);
        atomicAdd(dr + j + 2, xv.z);
        atomicAdd(dr + j + 3, xv.w);
      }
      if (col == 0) {
        enc_out[(size_t)token * K_EMB + idx] = 1.0f;
        atomicAdd(counts + idx, 1.0f);
      }
    }
  }
}

// ---------------------------------------------- cluster-size EMA + smoothing
__global__ __launch_bounds__(256) void ema_cs_kernel(
    const float* __restrict__ ecs_in, const float* __restrict__ counts,
    float* __restrict__ cs_out, float* __restrict__ cs_ws) {
  __shared__ float red[256];
  const int tid = threadIdx.x;
  float cs0[16];
  float s = 0.f;
#pragma unroll
  for (int i = 0; i < 16; ++i) {
    int k = tid + i * 256;
    cs0[i] = ecs_in[k] * DECAY + (1.0f - DECAY) * counts[k];
    s += cs0[i];
  }
  red[tid] = s;
  __syncthreads();
  for (int w = 128; w > 0; w >>= 1) {
    if (tid < w) red[tid] += red[tid + w];
    __syncthreads();
  }
  const float n = red[0];
  const float scale = n / (n + (float)K_EMB * EPS);
#pragma unroll
  for (int i = 0; i < 16; ++i) {
    int k = tid + i * 256;
    float cs = (cs0[i] + EPS) * scale;
    cs_out[k] = cs;
    cs_ws[k] = cs;
  }
}

// ------------------------------------------- ema_w update + codebook divide
__global__ __launch_bounds__(256) void ema_w_kernel(
    const float* __restrict__ ew_in, const float* __restrict__ dw,
    const float* __restrict__ cs, float* __restrict__ nw_out,
    float* __restrict__ ncb_out) {
  const int k = blockIdx.x;
  const int d = threadIdx.x;
  const size_t i = (size_t)k * DIM + d;
  float w = ew_in[i] * DECAY + (1.0f - DECAY) * dw[i];
  nw_out[i] = w;
  ncb_out[i] = w / cs[k];
}

extern "C" void kernel_launch(void* const* d_in, const int* in_sizes, int n_in,
                              void* d_out, int out_size, void* d_ws, size_t ws_size,
                              hipStream_t stream) {
  (void)in_sizes; (void)n_in; (void)out_size; (void)ws_size;
  const float* X   = (const float*)d_in[0];
  const float* CB  = (const float*)d_in[1];
  const float* EW  = (const float*)d_in[2];
  const float* ECS = (const float*)d_in[3];

  float* q_out   = (float*)d_out;
  float* enc_out = q_out   + (size_t)N_TOK * DIM;
  float* ncb_out = enc_out + (size_t)N_TOK * K_EMB;
  float* nw_out  = ncb_out + (size_t)K_EMB * DIM;
  float* cs_out  = nw_out  + (size_t)K_EMB * DIM;

  char* ws = (char*)d_ws;
  float* dw              = (float*)(ws + 0);                       // 4 MiB
  float* counts          = (float*)(ws + 4194304);                 // 16 KiB
  float* cs_ws           = (float*)(ws + 4194304 + 16384);         // 16 KiB
  float* enorm           = (float*)(ws + 4194304 + 32768);         // 16 KiB
  unsigned short* cb_hi  = (unsigned short*)(ws + 4194304 + 49152);          // 2 MiB
  unsigned short* cb_lo  = (unsigned short*)(ws + 4194304 + 49152 + 2097152);// 2 MiB

  // 1) zero one-hot encodings region (537 MB) and dw+counts scratch
  fill0_kernel<<<4096, 256, 0, stream>>>((float4*)enc_out,
                                         (size_t)N_TOK * K_EMB / 4);
  fill0_kernel<<<512, 256, 0, stream>>>((float4*)dw,
                                        (size_t)(K_EMB * DIM + K_EMB) / 4);

  // 2) codebook -> bf16 hi/lo + squared norms
  prep_codebook_kernel<<<K_EMB / 8, 256, 0, stream>>>(CB, cb_hi, cb_lo, enorm);

  // 3) fused distance/argmin/quantize/scatter (WMMA)
  vq_main_kernel<<<N_TOK / 256, 256, 0, stream>>>(X, CB, cb_hi, cb_lo, enorm,
                                                  q_out, enc_out, dw, counts);

  // 4) cluster-size EMA + laplace smoothing (single WG reduction)
  ema_cs_kernel<<<1, 256, 0, stream>>>(ECS, counts, cs_out, cs_ws);

  // 5) ema_w update + new codebook
  ema_w_kernel<<<K_EMB, 256, 0, stream>>>(EW, dw, cs_ws, nw_out, ncb_out);
}
